// SelfAttention_10857677324963
// MI455X (gfx1250) — matrix operand order repaired and checked by reading, not compiled, not run on hardware
//
#include <hip/hip_runtime.h>
#include <hip/hip_bf16.h>
#include <math.h>

// ---------------------------------------------------------------------------
// Fused causal self-attention for gfx1250 (MI455X), bf16 WMMA everywhere.
//   x:[4,2048,1024] f32 ; Wq/Wk/Wv:[1024,1024] f32 ; bq/bk/bv:[1024] f32
//   out:[4,2048,1024] f32
// Pipeline:
//   1) cvt_x      : x  -> xb  bf16            [8192][1024]
//   2) cvt_wT     : W  -> Wt  bf16 transposed [1024][1024]  (x3)
//   3) qkv_gemm   : Qb,Kb bf16 row-major ; V  bf16 transposed [B][H][S]
//   4) attn       : per (batch, 32-row q tile): WMMA scores -> LDS softmax
//                   (128KB dynamic LDS, CDNA5 320KB WGP) -> WMMA P@V
// ---------------------------------------------------------------------------

#define EMB 1024
#define HID 1024
#define SEQ 2048
#define NB  4

typedef __attribute__((ext_vector_type(8)))  __bf16 v8bf;
typedef __attribute__((ext_vector_type(16))) __bf16 v16bf;
typedef __attribute__((ext_vector_type(8)))  float  v8f;

static __device__ inline v16bf combine16(v8bf lo, v8bf hi) {
  v16bf a;
#pragma unroll
  for (int i = 0; i < 8; ++i) { a[i] = lo[i]; a[i + 8] = hi[i]; }
  return a;
}

// A operand 16x32 bf16 (ISA 7.12.2): lane l<16 -> row l, elems 0..7 = K hk*8+j,
// elems 8..15 = K 16+hk*8+j (hk = lane>>4). row0 points at (tile row 0, k0).
static __device__ inline v16bf load_a_frag(const __bf16* row0, size_t ld, int lane) {
  const int l16 = lane & 15, hk = (lane >> 4) & 1;
  const __bf16* p = row0 + (size_t)l16 * ld + (size_t)hk * 8;
  return combine16(*(const v8bf*)p, *(const v8bf*)(p + 16));
}

// B operand 32x16 bf16: lane%16 = output column n, elems 0..7 = K hk*8+j, elems 8..15 = K 16+hk*8+j.
// src is "N-major" (row n holds the K-dim contiguously), row0 at (n0, k0).
static __device__ inline v16bf load_b_frag(const __bf16* row0, size_t ld, int lane) {
  const int l16 = lane & 15, hk = (lane >> 4) & 1;
  const __bf16* p = row0 + (size_t)l16 * ld + (size_t)hk * 8;
  return combine16(*(const v8bf*)p, *(const v8bf*)(p + 16));
}

static __device__ inline v8f wmma_bf16(v16bf a, v16bf b, v8f c) {
  return __builtin_amdgcn_wmma_f32_16x16x32_bf16(false, a, false, b,
                                                 (short)0, c, false, false);
}

// ---------------------------------------------------------------------------
__global__ __launch_bounds__(256) void cvt_x_kernel(const float* __restrict__ x,
                                                    __bf16* __restrict__ xb, int n) {
  int i = blockIdx.x * 256 + threadIdx.x;
  if (i < n) xb[i] = (__bf16)x[i];
}

__global__ __launch_bounds__(256) void cvt_wT_kernel(const float* __restrict__ W,
                                                     __bf16* __restrict__ Wt) {
  int i = blockIdx.x * 256 + threadIdx.x;   // over EMB*HID, coalesced on W
  int e = i / HID, h = i % HID;
  Wt[(size_t)h * EMB + e] = (__bf16)W[i];
}

// ---------------------------------------------------------------------------
// C[8192,1024] = xb @ WtT + bias.  Block tile 128x128, 8 waves (4 along M, 2
// along N), each wave 32x64 = 2x4 wmma accumulators. Operands stream straight
// from global (all resident in 192MB L2).
// mode 0: store bf16 row-major [B*S][H]   (Q, K)
// mode 1: store bf16 transposed [B][H][S] (V)
__global__ __launch_bounds__(256) void qkv_gemm_kernel(
    const __bf16* __restrict__ xb, const __bf16* __restrict__ Wt,
    const float* __restrict__ bias, __bf16* __restrict__ dst, int mode) {
  const int tid  = threadIdx.x;
  const int lane = tid & 31, w = tid >> 5;
  const int wm = w & 3, wn = w >> 2;
  const int m_base = blockIdx.x * 128 + wm * 32;
  const int n_base = blockIdx.y * 128 + wn * 64;
  const int l16 = lane & 15, hk = (lane >> 4) & 1;

  v8f c[2][4] = {};
  for (int e0 = 0; e0 < EMB; e0 += 32) {
    v16bf a0 = load_a_frag(xb + (size_t)m_base * EMB + e0, EMB, lane);
    v16bf a1 = load_a_frag(xb + (size_t)(m_base + 16) * EMB + e0, EMB, lane);
#pragma unroll
    for (int nt = 0; nt < 4; ++nt) {
      v16bf b = load_b_frag(Wt + (size_t)(n_base + nt * 16) * EMB + e0, EMB, lane);
      c[0][nt] = wmma_bf16(a0, b, c[0][nt]);
      c[1][nt] = wmma_bf16(a1, b, c[1][nt]);
    }
  }
#pragma unroll
  for (int nt = 0; nt < 4; ++nt) {
    const int col = n_base + nt * 16 + l16;
    const float bv = bias[col];
#pragma unroll
    for (int mt = 0; mt < 2; ++mt) {
#pragma unroll
      for (int r = 0; r < 8; ++r) {
        const int row = m_base + mt * 16 + hk * 8 + r;
        const float v = c[mt][nt][r] + bv;
        if (mode == 0) {
          dst[(size_t)row * HID + col] = (__bf16)v;
        } else {
          const int bb = row >> 11, s = row & (SEQ - 1);
          dst[((size_t)bb * HID + col) * SEQ + s] = (__bf16)v;
        }
      }
    }
  }
}

// ---------------------------------------------------------------------------
// Attention: grid (qtile=64, batch=4). 256 threads. Dynamic LDS:
//   P[32][2048] bf16 (128KB) + rowmax[32] + rowsum[32] f32.
__global__ __launch_bounds__(256) void attn_kernel(
    const __bf16* __restrict__ Qb, const __bf16* __restrict__ Kb,
    const __bf16* __restrict__ VT, float* __restrict__ out) {
  extern __shared__ __bf16 smem[];
  __bf16* P      = smem;                      // [32][SEQ]
  float*  rowmax = (float*)(smem + 32 * SEQ);
  float*  rowsum = rowmax + 32;

  const int tid  = threadIdx.x;
  const int lane = tid & 31, w = tid >> 5;
  const int l16 = lane & 15, hk = (lane >> 4) & 1;
  const int qt = blockIdx.x, bb = blockIdx.y;
  const int q_base = qt * 32;
  const int nk = qt + 1;                      // causal: key tiles [0, nk)

  const __bf16* Qrow = Qb + ((size_t)bb * SEQ + q_base) * HID;
  const __bf16* Krow = Kb + (size_t)bb * SEQ * HID;

  // ---- Phase 1: scores -> P (bf16, scaled, causal-masked) ----
  const float scale = 0.03125f;               // 1/sqrt(1024)
  for (int kt = w; kt < nk; kt += 8) {        // wave-uniform partition
    const int k_base = kt * 32;
    v8f c[2][2] = {};
    for (int e0 = 0; e0 < HID; e0 += 32) {
      v16bf a0 = load_a_frag(Qrow + e0, HID, lane);
      v16bf a1 = load_a_frag(Qrow + (size_t)16 * HID + e0, HID, lane);
      v16bf b0 = load_b_frag(Krow + (size_t)k_base * HID + e0, HID, lane);
      v16bf b1 = load_b_frag(Krow + (size_t)(k_base + 16) * HID + e0, HID, lane);
      c[0][0] = wmma_bf16(a0, b0, c[0][0]);
      c[0][1] = wmma_bf16(a0, b1, c[0][1]);
      c[1][0] = wmma_bf16(a1, b0, c[1][0]);
      c[1][1] = wmma_bf16(a1, b1, c[1][1]);
    }
#pragma unroll
    for (int mt = 0; mt < 2; ++mt) {
#pragma unroll
      for (int nt = 0; nt < 2; ++nt) {
#pragma unroll
        for (int r = 0; r < 8; ++r) {
          const int mrow = mt * 16 + hk * 8 + r;
          const int kcol = nt * 16 + l16;
          float v = c[mt][nt][r] * scale;
          if (kt == qt && kcol > mrow) v = -INFINITY;   // diagonal tile mask
          P[(size_t)mrow * SEQ + k_base + kcol] = (__bf16)v;
        }
      }
    }
  }
  __syncthreads();

  // ---- Phase 2: row softmax over ncols = nk*32 (8 lanes per row) ----
  const int r   = tid >> 3;
  const int sub = tid & 7;
  const int ncols = nk * 32;
  float mx = -INFINITY;
  for (int cc = sub; cc < ncols; cc += 8)
    mx = fmaxf(mx, (float)P[(size_t)r * SEQ + cc]);
#pragma unroll
  for (int off = 4; off > 0; off >>= 1) mx = fmaxf(mx, __shfl_xor(mx, off, 8));
  if (sub == 0) rowmax[r] = mx;
  __syncthreads();
  mx = rowmax[r];
  float s = 0.f;
  for (int cc = sub; cc < ncols; cc += 8) {
    const size_t idx = (size_t)r * SEQ + cc;
    const float v = __expf((float)P[idx] - mx);
    P[idx] = (__bf16)v;                       // unnormalized weights
    s += v;
  }
#pragma unroll
  for (int off = 4; off > 0; off >>= 1) s += __shfl_xor(s, off, 8);
  if (sub == 0) rowsum[r] = s;
  __syncthreads();

  // ---- Phase 3: O = P @ V ; each wave owns 128 H-columns, 2 halves of 64 ----
  const __bf16* Vt = VT + (size_t)bb * HID * SEQ;   // [H][S]
  for (int half = 0; half < 2; ++half) {
    const int h_base = w * 128 + half * 64;
    v8f o[2][4] = {};
    for (int kt = 0; kt < nk; ++kt) {
      const int k0 = kt * 32;
      v16bf a0 = load_a_frag(P + k0, SEQ, lane);                    // LDS
      v16bf a1 = load_a_frag(P + (size_t)16 * SEQ + k0, SEQ, lane); // LDS
#pragma unroll
      for (int ht = 0; ht < 4; ++ht) {
        v16bf b = load_b_frag(Vt + (size_t)(h_base + ht * 16) * SEQ + k0, SEQ, lane);
        o[0][ht] = wmma_bf16(a0, b, o[0][ht]);
        o[1][ht] = wmma_bf16(a1, b, o[1][ht]);
      }
    }
#pragma unroll
    for (int ht = 0; ht < 4; ++ht) {
      const int h = h_base + ht * 16 + l16;
#pragma unroll
      for (int mt = 0; mt < 2; ++mt) {
#pragma unroll
        for (int rr = 0; rr < 8; ++rr) {
          const int mrow = mt * 16 + hk * 8 + rr;
          const float inv = 1.0f / rowsum[mrow];
          out[((size_t)bb * SEQ + q_base + mrow) * HID + h] = o[mt][ht][rr] * inv;
        }
      }
    }
  }
}

// ---------------------------------------------------------------------------
extern "C" void kernel_launch(void* const* d_in, const int* in_sizes, int n_in,
                              void* d_out, int out_size, void* d_ws, size_t ws_size,
                              hipStream_t stream) {
  (void)in_sizes; (void)n_in; (void)out_size; (void)ws_size;
  const float* x  = (const float*)d_in[0];
  const float* Wq = (const float*)d_in[1];
  const float* bq = (const float*)d_in[2];
  const float* Wk = (const float*)d_in[3];
  const float* bk = (const float*)d_in[4];
  const float* Wv = (const float*)d_in[5];
  const float* bv = (const float*)d_in[6];
  float* out = (float*)d_out;

  char* ws = (char*)d_ws;
  const size_t MB = 1024ull * 1024ull;
  __bf16* xb  = (__bf16*)(ws);                 // 16 MB  [8192][1024]
  __bf16* WqT = (__bf16*)(ws + 16 * MB);       //  2 MB  [1024][1024] (transposed)
  __bf16* WkT = (__bf16*)(ws + 18 * MB);
  __bf16* WvT = (__bf16*)(ws + 20 * MB);
  __bf16* Qb  = (__bf16*)(ws + 22 * MB);       // 16 MB  [8192][1024]
  __bf16* Kb  = (__bf16*)(ws + 38 * MB);       // 16 MB
  __bf16* VTb = (__bf16*)(ws + 54 * MB);       // 16 MB  [4][1024][2048]

  const int nx = NB * SEQ * EMB;
  cvt_x_kernel<<<nx / 256, 256, 0, stream>>>(x, xb, nx);
  cvt_wT_kernel<<<(EMB * HID) / 256, 256, 0, stream>>>(Wq, WqT);
  cvt_wT_kernel<<<(EMB * HID) / 256, 256, 0, stream>>>(Wk, WkT);
  cvt_wT_kernel<<<(EMB * HID) / 256, 256, 0, stream>>>(Wv, WvT);

  dim3 ggrid(NB * SEQ / 128, HID / 128);
  qkv_gemm_kernel<<<ggrid, 256, 0, stream>>>(xb, WqT, bq, Qb, 0);
  qkv_gemm_kernel<<<ggrid, 256, 0, stream>>>(xb, WkT, bk, Kb, 0);
  qkv_gemm_kernel<<<ggrid, 256, 0, stream>>>(xb, WvT, bv, VTb, 1);

  const size_t lds_bytes = (size_t)32 * SEQ * sizeof(__bf16) + 64 * sizeof(float);
  attn_kernel<<<dim3(SEQ / 32, NB), 256, lds_bytes, stream>>>(Qb, Kb, VTb, out);
}
